// HeteroGNN_28535762714971
// MI455X (gfx1250) — matrix operand order repaired and checked by reading, not compiled, not run on hardware
//
#include <hip/hip_runtime.h>
#include <hip/hip_bf16.h>

// ---------------------------------------------------------------------------
// HeteroGNN (2x SAGEConv + linear head) for gfx1250 (MI455X).
// Memory-bound (~2.6GB scatter traffic vs ~17 GFLOP GEMM). Strategy:
//  - one pass per aggregation (float4 gathers + f32 atomics)
//  - fused mean + dual-GEMM + bias + ReLU WMMA kernel; 8 waves/block share a
//    16xK A-slab staged in LDS via global_load_async_to_lds_b128 (ASYNCcnt),
//    converted f32->bf16 once cooperatively (v_cvt_pk_bf16_f32), fragments
//    then built purely from ds_load_b128 (no per-wave VALU conversion)
//  - weights pre-packed once per call into bf16 WMMA B-fragment layout
// ---------------------------------------------------------------------------

typedef __attribute__((ext_vector_type(16))) __bf16 v16bf;
typedef __attribute__((ext_vector_type(8)))  __bf16 v8bf;
typedef __attribute__((ext_vector_type(2)))  __bf16 v2bf;
typedef __attribute__((ext_vector_type(8)))  float  v8f;

#define HDIM 128

__device__ __forceinline__ unsigned short bf16bits(float f) {
  // round-to-nearest-even f32 -> bf16 (software fallback)
  unsigned u = __builtin_bit_cast(unsigned, f);
  u += 0x7FFFu + ((u >> 16) & 1u);
  return (unsigned short)(u >> 16);
}

__device__ __forceinline__ unsigned pk_bf16(float x, float y) {
#if __has_builtin(__builtin_amdgcn_cvt_pk_bf16_f32)
  v2bf p = __builtin_amdgcn_cvt_pk_bf16_f32(x, y);
  return __builtin_bit_cast(unsigned, p);
#else
  return (unsigned)bf16bits(x) | ((unsigned)bf16bits(y) << 16);
#endif
}

__global__ void zero_f32(float* __restrict__ p, long n) {
  long i = (long)blockIdx.x * blockDim.x + threadIdx.x;
  long stride = (long)gridDim.x * blockDim.x;
  for (; i < n; i += stride) p[i] = 0.0f;
}

__global__ void degree_count(const int* __restrict__ dst, float* __restrict__ cnt, int E) {
  int i = blockIdx.x * blockDim.x + threadIdx.x;
  int stride = gridDim.x * blockDim.x;
  for (; i < E; i += stride) atomicAdd(&cnt[dst[i]], 1.0f);
}

// agg[dst[e], :] += xsrc[src[e], :]   (float4 gathers, scalar f32 atomics)
__global__ void scatter_add(const float* __restrict__ xsrc, const int* __restrict__ src,
                            const int* __restrict__ dst, float* __restrict__ agg,
                            int E, int K) {
  int kq = K >> 2;
  long total = (long)E * kq;
  long i = (long)blockIdx.x * blockDim.x + threadIdx.x;
  long stride = (long)gridDim.x * blockDim.x;
  for (; i < total; i += stride) {
    long e = i / kq;
    int  f = (int)(i - e * kq) << 2;
    long s = src[e], d = dst[e];
    float4 v = *(const float4*)(xsrc + s * K + f);
    float* a = agg + d * K + f;
    atomicAdd(a + 0, v.x);
    atomicAdd(a + 1, v.y);
    atomicAdd(a + 2, v.z);
    atomicAdd(a + 3, v.w);
  }
}

// Pre-pack W[K,HDIM] f32 -> bf16 in WMMA B-fragment order:
//   P[((colTile*(K/32) + kt)*32 + lane)*16 + h]  =  bf16(W[k*HDIM + n])
//   with k = kt*32 + (h&7) + 8*(lane>>4) + 16*(h>>3), n = colTile*16 + (lane&15)
__global__ void pack_w_bf16(const float* __restrict__ W, unsigned short* __restrict__ P, int K) {
  int i = blockIdx.x * blockDim.x + threadIdx.x;
  int total = K * HDIM;
  if (i >= total) return;
  int h = i & 15;
  int lane = (i >> 4) & 31;
  int kt = (i >> 9) % (K >> 5);
  int colTile = i / ((K >> 5) << 9);
  int k = (kt << 5) + (h & 7) + ((lane >> 4) << 3) + ((h >> 3) << 4);
  int n = (colTile << 4) + (lane & 15);
  P[i] = bf16bits(W[k * HDIM + n]);
}

// Y = ReLU( (A1/max(cnt,1)) @ Wl + A2 @ Wr + bias ), one block (8 waves) per
// 16-row tile; wave w handles column tile w.
template <int K1, int K2>
__global__ __launch_bounds__(256) void sage_wmma(
    const float* __restrict__ A1, const float* __restrict__ cnt,
    const float* __restrict__ A2,
    const unsigned short* __restrict__ Pl,   // packed bf16 [HDIM/16][K1/32][32][16]
    const unsigned short* __restrict__ Pr,   // packed bf16 [HDIM/16][K2/32][32][16]
    const float* __restrict__ bias, float* __restrict__ Y) {
  __shared__ __align__(16) float  smF1[16 * K1];
  __shared__ __align__(16) float  smF2[16 * K2];
  __shared__ __align__(16) __bf16 smB1[16 * K1];
  __shared__ __align__(16) __bf16 smB2[16 * K2];

  const int tid = threadIdx.x;
  const int rowTile = blockIdx.x;

  // --- stage 16xK1 and 16xK2 A-slabs into LDS (async, 16B per lane per op)
  {
    const float* g1 = A1 + (long)rowTile * 16 * K1;
#pragma unroll
    for (int f = 0; f < 16 * K1; f += 1024) {
      int idx = f + tid * 4;
      unsigned lds = (unsigned)(unsigned long long)(smF1 + idx);
      unsigned long long ga = (unsigned long long)(g1 + idx);
      asm volatile("global_load_async_to_lds_b128 %0, %1, off" ::"v"(lds), "v"(ga) : "memory");
    }
    const float* g2 = A2 + (long)rowTile * 16 * K2;
#pragma unroll
    for (int f = 0; f < 16 * K2; f += 1024) {
      int idx = f + tid * 4;
      unsigned lds = (unsigned)(unsigned long long)(smF2 + idx);
      unsigned long long ga = (unsigned long long)(g2 + idx);
      asm volatile("global_load_async_to_lds_b128 %0, %1, off" ::"v"(lds), "v"(ga) : "memory");
    }
  }
  asm volatile("s_wait_asynccnt 0x0" ::: "memory");
  __syncthreads();

  // --- cooperative f32 -> bf16 conversion (once per block, not per wave);
  //     mean normalization fused for the aggregated branch
#pragma unroll
  for (int f0 = 0; f0 < 16 * K1; f0 += 1024) {
    int f = f0 + tid * 4;
    int row = f / K1;
    float rs = 1.0f / fmaxf(cnt[rowTile * 16 + row], 1.0f);
    float4 v = *(const float4*)(smF1 + f);
    unsigned* d = (unsigned*)(smB1 + f);
    d[0] = pk_bf16(v.x * rs, v.y * rs);
    d[1] = pk_bf16(v.z * rs, v.w * rs);
  }
#pragma unroll
  for (int f0 = 0; f0 < 16 * K2; f0 += 1024) {
    int f = f0 + tid * 4;
    float4 v = *(const float4*)(smF2 + f);
    unsigned* d = (unsigned*)(smB2 + f);
    d[0] = pk_bf16(v.x, v.y);
    d[1] = pk_bf16(v.z, v.w);
  }
  __syncthreads();

  const int wave = tid >> 5;  // column tile (0..7)
  const int lane = tid & 31;
  const int grp = lane >> 4;
  const int lrow = lane & 15;

  v8f c = {};

  // branch 1: mean(A1) @ Wl -- fragments are pure ds_load_b128 pairs
  {
    const __bf16* r1 = smB1 + lrow * K1;
#pragma unroll
    for (int kt = 0; kt < K1; kt += 32) {
      v8bf lo = *(const v8bf*)(r1 + kt + grp * 8);
      v8bf hi = *(const v8bf*)(r1 + kt + 16 + grp * 8);
      v16bf a = __builtin_shufflevector(lo, hi, 0, 1, 2, 3, 4, 5, 6, 7,
                                        8, 9, 10, 11, 12, 13, 14, 15);
      const v16bf b = *(const v16bf*)(Pl + (((wave * (K1 / 32) + (kt >> 5)) * 32 + lane) << 4));
      c = __builtin_amdgcn_wmma_f32_16x16x32_bf16(false, a, false, b, (short)0, c, false, false);
    }
  }
  // branch 2: A2 @ Wr
  {
    const __bf16* r2 = smB2 + lrow * K2;
#pragma unroll
    for (int kt = 0; kt < K2; kt += 32) {
      v8bf lo = *(const v8bf*)(r2 + kt + grp * 8);
      v8bf hi = *(const v8bf*)(r2 + kt + 16 + grp * 8);
      v16bf a = __builtin_shufflevector(lo, hi, 0, 1, 2, 3, 4, 5, 6, 7,
                                        8, 9, 10, 11, 12, 13, 14, 15);
      const v16bf b = *(const v16bf*)(Pr + (((wave * (K2 / 32) + (kt >> 5)) * 32 + lane) << 4));
      c = __builtin_amdgcn_wmma_f32_16x16x32_bf16(false, a, false, b, (short)0, c, false, false);
    }
  }

  const int n = wave * 16 + lrow;
  const float bn = bias[n];
#pragma unroll
  for (int r = 0; r < 8; ++r) {
    int m = rowTile * 16 + r + (grp << 3);
    float y = c[r] + bn;
    Y[(long)m * HDIM + n] = y > 0.0f ? y : 0.0f;
  }
}

// out[i] = sigmoid( dot(u2[mu[i]], Wlin[0:128]) + dot(s2[ms[i]], Wlin[128:256]) + blin )
__global__ void head_kernel(const float* __restrict__ u2, const float* __restrict__ s2,
                            const int* __restrict__ mu, const int* __restrict__ ms,
                            const float* __restrict__ Wlin, const float* __restrict__ blin,
                            float* __restrict__ out, int B) {
  int wave = (int)((blockIdx.x * (long)blockDim.x + threadIdx.x) >> 5);
  int lane = threadIdx.x & 31;
  if (wave >= B) return;
  long iu = (long)mu[wave] * HDIM;
  long is = (long)ms[wave] * HDIM;
  float acc = 0.0f;
#pragma unroll
  for (int t = 0; t < 4; ++t) {
    int j = lane + 32 * t;
    acc += u2[iu + j] * Wlin[j];
    acc += s2[is + j] * Wlin[HDIM + j];
  }
#pragma unroll
  for (int off = 16; off > 0; off >>= 1) acc += __shfl_xor(acc, off, 32);
  if (lane == 0) out[wave] = 1.0f / (1.0f + __expf(-(acc + blin[0])));
}

extern "C" void kernel_launch(void* const* d_in, const int* in_sizes, int n_in,
                              void* d_out, int out_size, void* d_ws, size_t ws_size,
                              hipStream_t stream) {
  const float* x_user   = (const float*)d_in[0];
  const float* x_seller = (const float*)d_in[1];
  const int*   src_buy  = (const int*)d_in[2];
  const int*   dst_buy  = (const int*)d_in[3];
  const int*   src_rev  = (const int*)d_in[4];
  const int*   dst_rev  = (const int*)d_in[5];
  const int*   mask_u   = (const int*)d_in[6];
  const int*   mask_s   = (const int*)d_in[7];
  const float* Wl1_buy  = (const float*)d_in[8];
  const float* Wr1_buy  = (const float*)d_in[9];
  const float* b1_buy   = (const float*)d_in[10];
  const float* Wl1_rev  = (const float*)d_in[11];
  const float* Wr1_rev  = (const float*)d_in[12];
  const float* b1_rev   = (const float*)d_in[13];
  const float* Wl2_buy  = (const float*)d_in[14];
  const float* Wr2_buy  = (const float*)d_in[15];
  const float* b2_buy   = (const float*)d_in[16];
  const float* Wl2_rev  = (const float*)d_in[17];
  const float* Wr2_rev  = (const float*)d_in[18];
  const float* b2_rev   = (const float*)d_in[19];
  const float* Wlin     = (const float*)d_in[20];
  const float* blin     = (const float*)d_in[21];
  float* out = (float*)d_out;

  const int D_U = 128, D_S = 64;
  const int N_USER   = in_sizes[0] / D_U;
  const int N_SELLER = in_sizes[1] / D_S;
  const int E = in_sizes[2];
  const int B = in_sizes[6];

  // ---- workspace layout: activations | packed weights | zero-init region
  float* ws = (float*)d_ws;
  size_t off = 0;
  float* s1 = ws + off; off += (size_t)N_SELLER * HDIM;
  float* u1 = ws + off; off += (size_t)N_USER * HDIM;
  float* s2 = ws + off; off += (size_t)N_SELLER * HDIM;
  float* u2 = ws + off; off += (size_t)N_USER * HDIM;

  unsigned short* pw = (unsigned short*)(ws + off);
  size_t poff = 0;
  unsigned short* P_l1b_l = pw + poff; poff += (size_t)D_U * HDIM;  // Wl1_buy  K=128
  unsigned short* P_l1b_r = pw + poff; poff += (size_t)D_S * HDIM;  // Wr1_buy  K=64
  unsigned short* P_l1r_l = pw + poff; poff += (size_t)D_S * HDIM;  // Wl1_rev  K=64
  unsigned short* P_l1r_r = pw + poff; poff += (size_t)D_U * HDIM;  // Wr1_rev  K=128
  unsigned short* P_l2b_l = pw + poff; poff += (size_t)HDIM * HDIM;
  unsigned short* P_l2b_r = pw + poff; poff += (size_t)HDIM * HDIM;
  unsigned short* P_l2r_l = pw + poff; poff += (size_t)HDIM * HDIM;
  unsigned short* P_l2r_r = pw + poff; poff += (size_t)HDIM * HDIM;
  off += (poff + 1) / 2;  // floats consumed by packed-weight region

  float* cnt_buy = ws + off; off += (size_t)N_SELLER;  // zero region start
  float* cnt_rev = ws + off; off += (size_t)N_USER;
  float* agg_s = ws + off; off += (size_t)N_SELLER * HDIM;
  float* agg_u = ws + off; off += (size_t)N_USER * HDIM;

  const int ZT = 256;
  long zero_all = (long)N_SELLER + N_USER + (long)N_SELLER * HDIM + (long)N_USER * HDIM;
  long zero_agg = (long)N_SELLER * HDIM + (long)N_USER * HDIM;

  // ---- pack weights to bf16 B-fragment layout (tiny, L2-resident afterwards)
  pack_w_bf16<<<dim3((D_U * HDIM + ZT - 1) / ZT), dim3(ZT), 0, stream>>>(Wl1_buy, P_l1b_l, D_U);
  pack_w_bf16<<<dim3((D_S * HDIM + ZT - 1) / ZT), dim3(ZT), 0, stream>>>(Wr1_buy, P_l1b_r, D_S);
  pack_w_bf16<<<dim3((D_S * HDIM + ZT - 1) / ZT), dim3(ZT), 0, stream>>>(Wl1_rev, P_l1r_l, D_S);
  pack_w_bf16<<<dim3((D_U * HDIM + ZT - 1) / ZT), dim3(ZT), 0, stream>>>(Wr1_rev, P_l1r_r, D_U);
  pack_w_bf16<<<dim3((HDIM * HDIM + ZT - 1) / ZT), dim3(ZT), 0, stream>>>(Wl2_buy, P_l2b_l, HDIM);
  pack_w_bf16<<<dim3((HDIM * HDIM + ZT - 1) / ZT), dim3(ZT), 0, stream>>>(Wr2_buy, P_l2b_r, HDIM);
  pack_w_bf16<<<dim3((HDIM * HDIM + ZT - 1) / ZT), dim3(ZT), 0, stream>>>(Wl2_rev, P_l2r_l, HDIM);
  pack_w_bf16<<<dim3((HDIM * HDIM + ZT - 1) / ZT), dim3(ZT), 0, stream>>>(Wr2_rev, P_l2r_r, HDIM);

  // ---- phase 0: clear counts + accumulators, compute degrees (shared by both layers)
  zero_f32<<<dim3((unsigned)((zero_all + ZT - 1) / ZT)), dim3(ZT), 0, stream>>>(cnt_buy, zero_all);
  degree_count<<<dim3((E + ZT - 1) / ZT), dim3(ZT), 0, stream>>>(dst_buy, cnt_buy, E);
  degree_count<<<dim3((E + ZT - 1) / ZT), dim3(ZT), 0, stream>>>(dst_rev, cnt_rev, E);

  // ---- layer 1 aggregation
  {
    long tb = (long)E * (D_U >> 2);
    scatter_add<<<dim3((unsigned)((tb + ZT - 1) / ZT)), dim3(ZT), 0, stream>>>(
        x_user, src_buy, dst_buy, agg_s, E, D_U);
    long tr = (long)E * (D_S >> 2);
    scatter_add<<<dim3((unsigned)((tr + ZT - 1) / ZT)), dim3(ZT), 0, stream>>>(
        x_seller, src_rev, dst_rev, agg_u, E, D_S);
  }
  // ---- layer 1 fused SAGE (WMMA)
  sage_wmma<128, 64><<<dim3(N_SELLER / 16), dim3(256), 0, stream>>>(
      agg_s, cnt_buy, x_seller, P_l1b_l, P_l1b_r, b1_buy, s1);
  sage_wmma<64, 128><<<dim3(N_USER / 16), dim3(256), 0, stream>>>(
      agg_u, cnt_rev, x_user, P_l1r_l, P_l1r_r, b1_rev, u1);

  // ---- layer 2 aggregation (reuse accumulators)
  zero_f32<<<dim3((unsigned)((zero_agg + ZT - 1) / ZT)), dim3(ZT), 0, stream>>>(agg_s, zero_agg);
  {
    long tb = (long)E * (HDIM >> 2);
    scatter_add<<<dim3((unsigned)((tb + ZT - 1) / ZT)), dim3(ZT), 0, stream>>>(
        u1, src_buy, dst_buy, agg_s, E, HDIM);
    scatter_add<<<dim3((unsigned)((tb + ZT - 1) / ZT)), dim3(ZT), 0, stream>>>(
        s1, src_rev, dst_rev, agg_u, E, HDIM);
  }
  // ---- layer 2 fused SAGE (WMMA)
  sage_wmma<128, 128><<<dim3(N_SELLER / 16), dim3(256), 0, stream>>>(
      agg_s, cnt_buy, s1, P_l2b_l, P_l2b_r, b2_buy, s2);
  sage_wmma<128, 128><<<dim3(N_USER / 16), dim3(256), 0, stream>>>(
      agg_u, cnt_rev, u1, P_l2r_l, P_l2r_r, b2_rev, u2);

  // ---- head: masked gather + concat + linear + sigmoid (one wave per sample)
  head_kernel<<<dim3((B + 7) / 8), dim3(256), 0, stream>>>(
      u2, s2, mask_u, mask_s, Wlin, blin, out, B);
}
